// Model_48043504173670
// MI455X (gfx1250) — matrix operand order repaired and checked by reading, not compiled, not run on hardware
//
#include <hip/hip_runtime.h>
#include <hip/hip_bf16.h>

typedef __attribute__((ext_vector_type(16))) _Float16 v16h;
typedef __attribute__((ext_vector_type(8)))  _Float16 h8;
typedef __attribute__((ext_vector_type(8)))  float    v8f;

#define BM 128
#define BN 64
#define BK 32
#define PADH 8   // row stride = 40 halves = 80B: 16B-aligned & bank-spread

__device__ __forceinline__ float silu_f(float x) { return x / (1.f + __expf(-x)); }
__device__ __forceinline__ float softplus_f(float x) {
  return (x > 20.f) ? x : log1pf(__expf(x));
}

// gfx1250 async global->LDS copy (16B per lane), tracked by ASYNCcnt.
__device__ __forceinline__ void async_b128(unsigned lds_off, const void* gaddr) {
  asm volatile("global_load_async_to_lds_b128 %0, %1, off"
               :: "v"(lds_off), "v"(gaddr) : "memory");
}
__device__ __forceinline__ void async_b128_o16(unsigned lds_off, const void* gaddr) {
  // offset:16 applies to both LDS and global addresses (ISA 15.18.3)
  asm volatile("global_load_async_to_lds_b128 %0, %1, off offset:16"
               :: "v"(lds_off), "v"(gaddr) : "memory");
}
__device__ __forceinline__ void wait_async0() {
  asm volatile("s_wait_asynccnt 0x0" ::: "memory");
}

// C[M,N] = act( A[M,K] @ W[N,K]^T + bias + add ), A/W in f16, accumulate f32.
// 128x64 block tile, 8 waves as 4(M) x 2(N); each wave: 32x32 out = 4 WMMAs/K-step.
// Tiles staged to LDS with GLOBAL_LOAD_ASYNC_TO_LDS_B128.
__global__ __launch_bounds__(256) void gemm_wmma_kernel(
    const _Float16* __restrict__ A, int lda,
    const _Float16* __restrict__ W, int ldw,
    float* __restrict__ C, int ldc,
    const float* __restrict__ bias,
    const float* __restrict__ add, int ldadd,
    _Float16* __restrict__ Ch,          // optional f16 mirror of C (same ldc)
    int M, int N, int K, int act)
{
  __shared__ alignas(16) _Float16 As[BM][BK + PADH];
  __shared__ alignas(16) _Float16 Bs[BN][BK + PADH];

  const int m0   = blockIdx.x * BM;
  const int n0   = blockIdx.y * BN;
  const int tid  = threadIdx.x;
  const int lane = tid & 31;
  const int wave = tid >> 5;
  const int wm   = wave >> 1;   // 0..3 -> M offset wm*32
  const int wn   = wave & 1;    // 0..1 -> N offset wn*32
  const int r    = lane & 15;
  const int hi   = lane >> 4;

  // LDS byte offsets (generic shared pointer carries LDS offset in low 32 bits)
  const unsigned asBase = (unsigned)(uintptr_t)&As[0][0];
  const unsigned bsBase = (unsigned)(uintptr_t)&Bs[0][0];
  const int rowStrB = (BK + PADH) * 2;   // 80 bytes

  // staging maps
  const int arow = tid >> 1;             // 0..127
  const int akb  = (tid & 1) * 16;       // halves: 0 or 16
  const int brow = tid >> 2;             // 0..63
  const int bkb  = (tid & 3) * 8;        // halves: 0,8,16,24
  const unsigned aLds = asBase + (unsigned)(arow * rowStrB + akb * 2);
  const unsigned bLds = bsBase + (unsigned)(brow * rowStrB + bkb * 2);
  const int bn = n0 + brow;

  v8f acc[2][2] = {};

  auto loadA = [&](int row) -> v16h {
    union { v16h v; h8 h[2]; } u;
    u.h[0] = *(const h8*)&As[row][hi * 8];        // K = hi*8   .. +7
    u.h[1] = *(const h8*)&As[row][16 + hi * 8];   // K = 16+hi*8.. +7
    return u.v;
  };
  auto loadB = [&](int row) -> v16h {
    union { v16h v; h8 h[2]; } u;
    u.h[0] = *(const h8*)&Bs[row][hi * 8];        // K = hi*8   .. +7
    u.h[1] = *(const h8*)&Bs[row][16 + hi * 8];   // K = 16+hi*8.. +7
    return u.v;
  };

  for (int k0 = 0; k0 < K; k0 += BK) {
    const bool kfull = (k0 + BK <= K);
    const bool nfull = (n0 + BN <= N);

    if (kfull) {
      const _Float16* gA = A + (size_t)(m0 + arow) * lda + k0 + akb;
      async_b128(aLds, gA);
      async_b128_o16(aLds, gA);          // halves +8..+15
    }
    if (kfull && nfull) {
      const _Float16* gB = W + (size_t)bn * ldw + k0 + bkb;
      async_b128(bLds, gB);
    }
    if (!kfull || !nfull) {
      // zero-prefill irregular tiles, then guarded 16B-granule asyncs
      h8 z = {};
      if (!kfull) {
        *(h8*)&As[arow][akb]     = z;
        *(h8*)&As[arow][akb + 8] = z;
      }
      if (!kfull || !nfull) *(h8*)&Bs[brow][bkb] = z;
      __syncthreads();   // zeros visible before async writes
      if (!kfull) {
        const _Float16* gA = A + (size_t)(m0 + arow) * lda + k0 + akb;
        if (k0 + akb + 8  <= K) async_b128(aLds, gA);
        if (k0 + akb + 16 <= K) async_b128_o16(aLds, gA);
      }
      if (bn < N && (k0 + bkb + 8 <= K) && !(kfull && nfull)) {
        const _Float16* gB = W + (size_t)bn * ldw + k0 + bkb;
        async_b128(bLds, gB);
      }
    }
    wait_async0();
    __syncthreads();

    v16h a0 = loadA(wm * 32 + r);
    v16h a1 = loadA(wm * 32 + 16 + r);
    v16h b0 = loadB(wn * 32 + r);
    v16h b1 = loadB(wn * 32 + 16 + r);

    acc[0][0] = __builtin_amdgcn_wmma_f32_16x16x32_f16(false, a0, false, b0,
                                                       (short)0, acc[0][0], false, false);
    acc[0][1] = __builtin_amdgcn_wmma_f32_16x16x32_f16(false, a0, false, b1,
                                                       (short)0, acc[0][1], false, false);
    acc[1][0] = __builtin_amdgcn_wmma_f32_16x16x32_f16(false, a1, false, b0,
                                                       (short)0, acc[1][0], false, false);
    acc[1][1] = __builtin_amdgcn_wmma_f32_16x16x32_f16(false, a1, false, b1,
                                                       (short)0, acc[1][1], false, false);
    __syncthreads();
  }

  // epilogue: C/D layout: elem v -> row hi*8+v, col r (per 16x16 tile)
  #pragma unroll
  for (int ni = 0; ni < 2; ++ni) {
    int col = n0 + wn * 32 + ni * 16 + r;
    if (col < N) {
      float bv = bias ? bias[col] : 0.f;
      #pragma unroll
      for (int mi = 0; mi < 2; ++mi) {
        #pragma unroll
        for (int v = 0; v < 8; ++v) {
          int row = m0 + wm * 32 + mi * 16 + hi * 8 + v;
          float o = acc[mi][ni][v] + bv;
          if (add)  o += add[(size_t)row * ldadd + col];
          if (act == 1)      o = silu_f(o);
          else if (act == 2) o = softplus_f(o);
          C[(size_t)row * ldc + col] = o;
          if (Ch) Ch[(size_t)row * ldc + col] = (_Float16)o;
        }
      }
    }
  }
}

// f32 -> f16 copy
__global__ void cvt_f16_kernel(const float* __restrict__ s,
                               _Float16* __restrict__ d, int n)
{
  int t = blockIdx.x * blockDim.x + threadIdx.x;
  if (t < n) d[t] = (_Float16)s[t];
}

// depthwise causal conv (width 4) + bias + SiLU; writes f32 + f16 mirror.
__global__ void conv_silu_kernel(const float* __restrict__ xz, int ldxz,
                                 const float* __restrict__ w,
                                 const float* __restrict__ b,
                                 float* __restrict__ out,
                                 _Float16* __restrict__ outh,
                                 int di, int L, int total)
{
  int t = blockIdx.x * blockDim.x + threadIdx.x;
  if (t >= total) return;
  int d  = t % di;
  int bl = t / di;       // b*L + l
  int l  = bl % L;
  float acc = b[d];
  const float* wd = w + d * 4;
  #pragma unroll
  for (int j = 0; j < 4; ++j) {
    int li = l - 3 + j;
    if (li >= 0) acc += wd[j] * xz[(size_t)(bl - 3 + j) * ldxz + d];
  }
  float r = silu_f(acc);
  out[(size_t)bl * di + d]  = r;
  outh[(size_t)bl * di + d] = (_Float16)r;
}

// fused selective scan: thread = (b, d, s); h kept in register; y reduced
// over the 16 states with wave32 shfl_xor (stays within a 16-lane group).
__global__ void scan_kernel(const float* __restrict__ dt,
                            const float* __restrict__ xc,
                            const float* __restrict__ dbc, int ldd, int boff, int coff,
                            const float* __restrict__ A_log,
                            const float* __restrict__ Dp,
                            float* __restrict__ y,
                            int di, int L, int total)
{
  int t = blockIdx.x * blockDim.x + threadIdx.x;
  if (t >= total) return;
  int s  = t & 15;
  int dd = (t >> 4) % di;
  int b  = (t >> 4) / di;
  float a  = -__expf(A_log[dd * 16 + s]);
  float Dv = Dp[dd];
  float h  = 0.f;
  size_t base = (size_t)b * L;
  for (int l = 0; l < L; ++l) {
    size_t rowi = base + l;
    float dtv = dt[rowi * di + dd];
    float xv  = xc[rowi * di + dd];
    float bm  = dbc[rowi * ldd + boff + s];
    float cm  = dbc[rowi * ldd + coff + s];
    h = __expf(dtv * a) * h + (dtv * xv) * bm;
    float acc = h * cm;
    acc += __shfl_xor(acc, 1, 32);
    acc += __shfl_xor(acc, 2, 32);
    acc += __shfl_xor(acc, 4, 32);
    acc += __shfl_xor(acc, 8, 32);
    if (s == 0) y[rowi * di + dd] = acc + xv * Dv;
  }
}

// y *= silu(z); writes f32 (in place) + f16 mirror.
__global__ void gate_kernel(float* __restrict__ y,
                            const float* __restrict__ xz, int ldxz,
                            _Float16* __restrict__ yh,
                            int di, int total)
{
  int t = blockIdx.x * blockDim.x + threadIdx.x;
  if (t >= total) return;
  int d = t % di;
  size_t row = (size_t)(t / di);
  float r = y[t] * silu_f(xz[row * ldxz + di + d]);
  y[t]  = r;
  yh[t] = (_Float16)r;
}

// ------------------------- host orchestration -------------------------

static inline void gemm_launch(const _Float16* A, int lda, const _Float16* W, int ldw,
                               float* C, int ldc, const float* bias,
                               const float* add, int ldadd, _Float16* Ch,
                               int M, int N, int K, int act, hipStream_t s)
{
  dim3 grid(M / BM, (N + BN - 1) / BN);
  gemm_wmma_kernel<<<grid, 256, 0, s>>>(A, lda, W, ldw, C, ldc, bias, add, ldadd, Ch,
                                        M, N, K, act);
}

static inline void cvt_launch(const float* s, _Float16* d, int n, hipStream_t st) {
  cvt_f16_kernel<<<(n + 255) / 256, 256, 0, st>>>(s, d, n);
}

struct MambaWS {
  float *xz, *xc, *dbc, *dt, *y;
  _Float16 *xch, *dbch, *yh;
  _Float16 *wA, *wX, *wDT, *wO;   // f16 weight staging
};

static void run_mamba(const _Float16* xinh, int d_model, void* const* d_in, int p,
                      float* out, const float* add_src, _Float16* outh,
                      const MambaWS& w, hipStream_t s)
{
  const int Bsz = 8, L = 512, M = Bsz * L;
  const int di  = 2 * d_model;
  const int dtr = (d_model + 15) / 16;
  const int ldd = dtr + 32;

  const float* in_proj  = (const float*)d_in[p + 0];
  const float* conv_w   = (const float*)d_in[p + 1];
  const float* conv_b   = (const float*)d_in[p + 2];
  const float* x_proj   = (const float*)d_in[p + 3];
  const float* dt_w     = (const float*)d_in[p + 4];
  const float* dt_b     = (const float*)d_in[p + 5];
  const float* A_log    = (const float*)d_in[p + 6];
  const float* Dp       = (const float*)d_in[p + 7];
  const float* out_proj = (const float*)d_in[p + 8];

  // stage weights to f16
  cvt_launch(in_proj,  w.wA,  2 * di * d_model, s);
  cvt_launch(x_proj,   w.wX,  ldd * di,         s);
  cvt_launch(dt_w,     w.wDT, di * dtr,         s);
  cvt_launch(out_proj, w.wO,  d_model * di,     s);

  // xz = x @ in_proj^T   (M x 2di), f32 (conv/gate consume f32)
  gemm_launch(xinh, d_model, w.wA, d_model, w.xz, 2 * di,
              nullptr, nullptr, 0, nullptr, M, 2 * di, d_model, 0, s);
  // xc = silu(conv(xz[:, :di]) + conv_b)  -> f32 + f16
  int totc = M * di;
  conv_silu_kernel<<<(totc + 255) / 256, 256, 0, s>>>(w.xz, 2 * di, conv_w, conv_b,
                                                      w.xc, w.xch, di, L, totc);
  // dbc = xc @ x_proj^T  (M x (dtr+32)) -> f32 (scan) + f16 (dt gemm)
  gemm_launch(w.xch, di, w.wX, di, w.dbc, ldd,
              nullptr, nullptr, 0, w.dbch, M, ldd, di, 0, s);
  // dt = softplus(dbc[:, :dtr] @ dt_w^T + dt_b)   (M x di)
  gemm_launch(w.dbch, ldd, w.wDT, dtr, w.dt, di,
              dt_b, nullptr, 0, nullptr, M, di, dtr, 2, s);
  // fused selective scan -> y (includes + xc * D)
  int tots = Bsz * di * 16;
  scan_kernel<<<(tots + 255) / 256, 256, 0, s>>>(w.dt, w.xc, w.dbc, ldd, dtr, dtr + 16,
                                                 A_log, Dp, w.y, di, L, tots);
  // y *= silu(z)  -> f32 + f16
  gate_kernel<<<(totc + 255) / 256, 256, 0, s>>>(w.y, w.xz, 2 * di, w.yh, di, totc);
  // out = y @ out_proj^T (+ add_src)
  gemm_launch(w.yh, di, w.wO, di, out, d_model,
              nullptr, add_src, d_model, outh, M, d_model, di, 0, s);
}

extern "C" void kernel_launch(void* const* d_in, const int* in_sizes, int n_in,
                              void* d_out, int out_size, void* d_ws, size_t ws_size,
                              hipStream_t stream)
{
  (void)in_sizes; (void)n_in; (void)out_size; (void)ws_size;
  const int M = 4096;  // B*L = 8*512
  const float* x = (const float*)d_in[0];

  // input index map (dict order): 0=x, 1..9=m1p, 10..18=m2p, 19..27=m3p,
  // 28..36=m4p, 37=lin2_w, 38=lin2_b, 39=lin3_w, 40=lin3_b, 41=lin4_w, 42=lin4_b
  const float* lin2_w = (const float*)d_in[37];
  const float* lin2_b = (const float*)d_in[38];
  const float* lin3_w = (const float*)d_in[39];
  const float* lin3_b = (const float*)d_in[40];
  const float* lin4_w = (const float*)d_in[41];
  const float* lin4_b = (const float*)d_in[42];
  float* out = (float*)d_out;

  // ---- fp32 workspace ----
  float* ws = (float*)d_ws;
  size_t off = 0;
  MambaWS w;
  w.xz  = ws + off; off += (size_t)M * 1024;  // max 2*di
  w.xc  = ws + off; off += (size_t)M * 512;   // max di
  w.dbc = ws + off; off += (size_t)M * 48;    // max dtr+32
  w.dt  = ws + off; off += (size_t)M * 512;
  w.y   = ws + off; off += (size_t)M * 512;
  float* x4  = ws + off; off += (size_t)M * 256;
  float* xl2 = ws + off; off += (size_t)M * 128;
  float* o1  = ws + off; off += (size_t)M * 128;
  float* xl3 = ws + off; off += (size_t)M * 256;

  // ---- f16 workspace (16B-aligned: all block sizes are multiples of 8 halves) ----
  _Float16* hb = (_Float16*)(ws + off);
  size_t ho = 0;
  _Float16* xh   = hb + ho; ho += (size_t)M * 256;
  w.xch  = hb + ho; ho += (size_t)M * 512;
  w.dbch = hb + ho; ho += (size_t)M * 48;
  w.yh   = hb + ho; ho += (size_t)M * 512;
  _Float16* x4h  = hb + ho; ho += (size_t)M * 256;
  _Float16* xl2h = hb + ho; ho += (size_t)M * 128;
  _Float16* o1h  = hb + ho; ho += (size_t)M * 128;
  _Float16* xl3h = hb + ho; ho += (size_t)M * 256;
  w.wA  = hb + ho; ho += (size_t)1024 * 256;  // max in_proj
  w.wX  = hb + ho; ho += (size_t)48 * 512;    // max x_proj
  w.wDT = hb + ho; ho += (size_t)512 * 16;    // max dt_w
  w.wO  = hb + ho; ho += (size_t)256 * 512;   // max out_proj
  _Float16* l2h = hb + ho; ho += (size_t)128 * 256;
  _Float16* l3h = hb + ho; ho += (size_t)256 * 128;
  _Float16* l4h = hb + ho; ho += (size_t)96 * 512;

  cvt_launch(x, xh, M * 256, stream);

  // x3 = mamba3(x); x4 = mamba4(x) + x3
  run_mamba(xh, 256, d_in, 19, x4, nullptr, x4h, w, stream);
  run_mamba(xh, 256, d_in, 28, x4, x4, x4h, w, stream);
  // xl2 = x @ lin2^T + b
  cvt_launch(lin2_w, l2h, 128 * 256, stream);
  gemm_launch(xh, 256, l2h, 256, xl2, 128, lin2_b, nullptr, 0, xl2h,
              M, 128, 256, 0, stream);
  // o1 = mamba1(xl2) + xl2 ; o1 = mamba2(xl2) + o1  (= x1 + x_res2 + x2)
  run_mamba(xl2h, 128, d_in, 1, o1, xl2, o1h, w, stream);
  run_mamba(xl2h, 128, d_in, 10, o1, o1, o1h, w, stream);
  // xl3 = o1 @ lin3^T + b + x
  cvt_launch(lin3_w, l3h, 256 * 128, stream);
  gemm_launch(o1h, 128, l3h, 128, xl3, 256, lin3_b, x, 256, xl3h,
              M, 256, 128, 0, stream);
  // out = concat([xl3, x4]) @ lin4^T + b  == xl3 @ Wl^T + b  +  x4 @ Wr^T
  cvt_launch(lin4_w, l4h, 96 * 512, stream);
  gemm_launch(xl3h, 256, l4h, 512, out, 96, lin4_b, nullptr, 0, nullptr,
              M, 96, 256, 0, stream);
  gemm_launch(x4h, 256, l4h + 256, 512, out, 96, nullptr, out, 96, nullptr,
              M, 96, 256, 0, stream);
}
